// MRU_55619826483854
// MI455X (gfx1250) — matrix operand order repaired and checked by reading, not compiled, not run on hardware
//
#include <hip/hip_runtime.h>
#include <math.h>

// MRU scan for MI455X (gfx1250, wave32, WMMA + TDM).
// Phase 0: repack weights/activations to bf16 once (L2-resident).
// Phase 1: [mlp;gpre] = flat @ [W_mlp;Wg_flat]^T + bias as bf16 WMMA GEMM
//          (M=16384, N=1024, K=2560). B tiles staged to LDS by the Tensor
//          Data Mover (tensor_load_to_lds, double-buffered, s_wait_tensorcnt),
//          shared by all 8 waves of the block. mlp -> d_out, gpre -> ws.
// Phase 2: single persistent workgroup runs the sequential scan; per-step
//          attn @ Wg_attn^T uses v_wmma_f32_16x16x32_bf16 on packed bf16 Wa.

typedef __attribute__((ext_vector_type(16))) __bf16        v16bf;
typedef __attribute__((ext_vector_type(8)))  __bf16        v8bf;
typedef __attribute__((ext_vector_type(8)))  float         v8f;
typedef __attribute__((ext_vector_type(4)))  unsigned int  v4u;
typedef __attribute__((ext_vector_type(8)))  int           v8i;
typedef __attribute__((ext_vector_type(4)))  int           v4i;

#define Bv 8
#define Tv 2048
#define Dv 512
#define Wv 5
#define KDIM (Wv * Dv)        // 2560 flat length
#define GROW ((Wv + 1) * Dv)  // 3072 W_gate row stride
#define NTOT (2 * Dv)         // 1024 fused output columns

// ---- workspace layout (bytes) ----
#define WS_GPRE 0u            // (B*T) x 512 fp32 gate preactivation   32 MB
#define WS_WC   33554432u     // 1024 x 2560 bf16 packed [W_mlp;Wg_f]   5 MB
#define WS_WA   38797312u     // 512 x 512 bf16 Wg_attn              0.5 MB
#define WS_ABF  39321600u     // (B*T) x 512 bf16 outputs             16 MB

// ---------------------------------------------------------------- Phase 0 ---
__global__ __launch_bounds__(256) void pack_wc(const float* __restrict__ W_mlp,
                                               const float* __restrict__ W_gate,
                                               __bf16* __restrict__ Wc) {
  const long i = (long)blockIdx.x * 256 + threadIdx.x;   // 1024*2560 exact
  const int n = (int)(i / KDIM), k = (int)(i % KDIM);
  const float v = (n < Dv) ? W_mlp[(long)n * KDIM + k]
                           : W_gate[(long)(n - Dv) * GROW + k];
  Wc[i] = (__bf16)v;
}

__global__ __launch_bounds__(256) void pack_wa(const float* __restrict__ W_gate,
                                               __bf16* __restrict__ Wa) {
  const int i = blockIdx.x * 256 + threadIdx.x;          // 512*512 exact
  const int n = i >> 9, k = i & (Dv - 1);
  Wa[i] = (__bf16)W_gate[(long)n * GROW + KDIM + k];
}

__global__ __launch_bounds__(256) void pack_a(const float* __restrict__ outputs,
                                              __bf16* __restrict__ Abf) {
  const long i = (long)blockIdx.x * 256 + threadIdx.x;   // B*T*D exact
  Abf[i] = (__bf16)outputs[i];
}

// ------------------------------------------------- TDM tile load (D# build) -
// Stage a 2-D bf16 tile [rows=64][cols=32] from Wc (row stride KDIM elements)
// into contiguous LDS (64 rows x 64 bytes). Descriptor per CDNA5 ISA §8.3/8.4.
static __device__ __forceinline__ void tdm_load_b_tile(const __bf16* gsrc,
                                                       unsigned lds_off) {
  const unsigned long long ga = (unsigned long long)(size_t)gsrc;
  v4u g0;
  g0[0] = 1u;                                        // count=1, user mode
  g0[1] = lds_off;                                   // LDS byte address
  g0[2] = (unsigned)(ga & 0xffffffffu);              // global_addr[31:0]
  g0[3] = (unsigned)((ga >> 32) & 0x01ffffffu)       // global_addr[56:32]
          | (2u << 30);                              // type=2 (image)
  v8i g1;
  g1[0] = 1 << 16;                                   // data_size=1 (2 bytes)
  g1[1] = (KDIM & 0xffff) << 16;                     // tensor_dim0 lo16
  g1[2] = ((KDIM >> 16) & 0xffff) | ((NTOT & 0xffff) << 16);  // d0 hi | d1 lo
  g1[3] = ((NTOT >> 16) & 0xffff) | (32 << 16);      // d1 hi | tile_dim0=32
  g1[4] = 64;                                        // tile_dim1=64, tile_dim2=0
  g1[5] = KDIM;                                      // tensor_dim0_stride lo32
  g1[6] = 0;                                         // stride hi | d1_stride lo
  g1[7] = 0;
  const v4i z4 = {0, 0, 0, 0};
#if __clang_major__ >= 23
  const v8i z8 = {0, 0, 0, 0, 0, 0, 0, 0};
  __builtin_amdgcn_tensor_load_to_lds(g0, g1, z4, z4, z8, 0);
#else
  __builtin_amdgcn_tensor_load_to_lds(g0, g1, z4, z4, 0);
#endif
}

// ---------------------------------------------------------------- Phase 1 ---
__global__ __launch_bounds__(256) void mru_phase1(
    const __bf16* __restrict__ Abf,   // (B*T) x 512 bf16
    const __bf16* __restrict__ Wc,    // 1024 x 2560 bf16
    const float* __restrict__ b_mlp, const float* __restrict__ b_gate,
    float* __restrict__ mlp_out,      // d_out, (B*T) x 512
    float* __restrict__ gpre_out)     // ws,    (B*T) x 512
{
  __shared__ __bf16 sB[2][64 * 32];   // double-buffered 4 KB B tiles [n][k]

  const int lane = threadIdx.x & 31;
  const int wv   = threadIdx.x >> 5;  // 8 waves -> 8 x 16 M rows
  const int g    = lane >> 4;
  const int lrow = lane & 15;

  const int mBase = blockIdx.y * 128 + wv * 16;
  const int n0    = blockIdx.x * 64;

  // Per-lane A row: flat(b,t) = contiguous span of Abf, zero-padded past lim.
  const int m = mBase + lrow;
  const int b = m / Tv;
  const int t = m % Tv;
  const int start = (t - 2 < 0) ? 0 : t - 2;
  const int last  = (t + 2 > Tv - 1) ? Tv - 1 : t + 2;
  const int lim   = (last - start + 1) * Dv;          // multiple of 512
  const __bf16* __restrict__ arow = Abf + ((long)b * Tv + start) * Dv;

  if (wv == 0)
    tdm_load_b_tile(Wc + (long)n0 * KDIM, (unsigned)(size_t)&sB[0][0]);
  __builtin_amdgcn_s_wait_tensorcnt(0);
  __syncthreads();

  v8f acc[4] = {};

  for (int kk = 0; kk < KDIM / 32; ++kk) {
    const int kBase = kk * 32;
    const int cur = kk & 1;
    if (wv == 0 && kk + 1 < KDIM / 32)
      tdm_load_b_tile(Wc + (long)n0 * KDIM + (kBase + 32),
                      (unsigned)(size_t)&sB[cur ^ 1][0]);

    // A tile 16x32 bf16: lane lrow = M row; K(e) = g*16 + e
    v16bf a;
#pragma unroll
    for (int h = 0; h < 2; ++h) {
      const int k0 = kBase + g * 16 + h * 8;  // 8-elem chunk: all-in or all-out
      if (k0 < lim) {
        const v8bf x = *(const v8bf*)(arow + k0);
#pragma unroll
        for (int e = 0; e < 8; ++e) a[h * 8 + e] = x[e];
      } else {
#pragma unroll
        for (int e = 0; e < 8; ++e) a[h * 8 + e] = (__bf16)0.0f;
      }
    }

    // B from LDS: lane lrow = column n; 16 contiguous K per lane group
#pragma unroll
    for (int nt = 0; nt < 4; ++nt) {
      const v16bf bm = *(const v16bf*)&sB[cur][(nt * 16 + lrow) * 32 + g * 16];
      acc[nt] = __builtin_amdgcn_wmma_f32_16x16x32_bf16(
          false, a, false, bm, (short)0, acc[nt], false, false);
    }

    __builtin_amdgcn_s_wait_tensorcnt(0);   // tile kk+1 landed (issuing wave)
    __syncthreads();                        // visible to all waves
  }

  // C/D layout: VGPR r holds (M = 8*g + r, N = lrow)
#pragma unroll
  for (int nt = 0; nt < 4; ++nt) {
    const int n = n0 + nt * 16 + lrow;
    const float bias = (n < Dv) ? b_mlp[n] : b_gate[n - Dv];
#pragma unroll
    for (int r = 0; r < 8; ++r) {
      const long mr = mBase + 8 * g + r;
      const float v = acc[nt][r] + bias;
      if (n < Dv) mlp_out[mr * Dv + n] = v;
      else        gpre_out[mr * Dv + (n - Dv)] = v;
    }
  }
}

// ---------------------------------------------------------------- Phase 2 ---
__global__ __launch_bounds__(512) void mru_phase2(
    const float*  __restrict__ outputs,
    const __bf16* __restrict__ Wa,    // 512 x 512 bf16 Wg_attn
    const float*  __restrict__ gpre,  // ws:    (B*T) x 512
    float* __restrict__ out)          // d_out: holds mlp_all, rewritten in place
{
  __shared__ float  s_state[Bv][Dv];
  __shared__ float  s_attn[Bv][Dv];
  __shared__ __bf16 s_attnb[16][Dv];  // attn padded to M=16 for WMMA A
  __shared__ float  s_scores[Bv][Wv];
  __shared__ float  s_prob[Bv][Wv];

  const int tid  = threadIdx.x;
  const int lane = tid & 31;
  const int wv   = tid >> 5;          // 16 waves
  const int g    = lane >> 4;
  const int lrow = lane & 15;
  const float scale = 0.044194173824159216f;  // 1/sqrt(512)

  for (int i = tid; i < Bv * Dv; i += 512) {
    s_state[i >> 9][i & (Dv - 1)] = 0.0f;
    s_attnb[8 + (i >> 9)][i & (Dv - 1)] = (__bf16)0.0f;
  }
  __syncthreads();

  for (int t = 0; t < Tv; ++t) {
    const int start  = (t - 2 < 0) ? 0 : t - 2;
    const int last   = (t + 2 > Tv - 1) ? Tv - 1 : t + 2;
    const int nvalid = last - start + 1;

    // scores[b][w] = scale * <state_b, winf_{b,w}> (padded slots stay 0)
    for (int p = wv; p < Bv * Wv; p += 16) {
      const int bb = p / Wv, w = p % Wv;
      float s = 0.0f;
      if (w < nvalid) {
        const float* row = outputs + ((long)bb * Tv + start + w) * Dv;
        for (int d = lane; d < Dv; d += 32) s += s_state[bb][d] * row[d];
      }
      for (int off = 16; off > 0; off >>= 1) s += __shfl_down(s, off, 32);
      if (lane == 0) s_scores[bb][w] = s * scale;
    }
    __syncthreads();

    if (tid < Bv) {
      float mx = s_scores[tid][0];
      for (int w = 1; w < Wv; ++w) mx = fmaxf(mx, s_scores[tid][w]);
      float ex[Wv]; float sum = 0.0f;
      for (int w = 0; w < Wv; ++w) { ex[w] = expf(s_scores[tid][w] - mx); sum += ex[w]; }
      const float inv = 1.0f / sum;
      for (int w = 0; w < Wv; ++w) s_prob[tid][w] = ex[w] * inv;
    }
    __syncthreads();

    for (int i = tid; i < Bv * Dv; i += 512) {
      const int bb = i >> 9, d = i & (Dv - 1);
      float a = 0.0f;
      for (int w = 0; w < Wv; ++w)
        if (w < nvalid)
          a += s_prob[bb][w] * outputs[((long)bb * Tv + start + w) * Dv + d];
      s_attn[bb][d]  = a;
      s_attnb[bb][d] = (__bf16)a;
    }
    __syncthreads();

    // attnW = attn @ Wg_attn^T : M=16(pad), N=512, K=512; wave owns 32 cols
    const int nW = wv * 32;
    v8f acc[2] = {};
    for (int kk = 0; kk < Dv / 32; ++kk) {
      const int kBase = kk * 32;
      v16bf a;
#pragma unroll
      for (int e = 0; e < 16; ++e)
        a[e] = s_attnb[lrow][kBase + g * 16 + e];
#pragma unroll
      for (int nt = 0; nt < 2; ++nt) {
        const int n = nW + nt * 16 + lrow;
        const v16bf bm = *(const v16bf*)(Wa + (long)n * Dv + kBase + g * 16);
        acc[nt] = __builtin_amdgcn_wmma_f32_16x16x32_bf16(
            false, a, false, bm, (short)0, acc[nt], false, false);
      }
    }

    // combine: real batches are M=0..7 -> lane group g==0 (M = r = b)
    if (g == 0) {
#pragma unroll
      for (int nt = 0; nt < 2; ++nt) {
        const int n = nW + nt * 16 + lrow;
#pragma unroll
        for (int r = 0; r < 8; ++r) {
          const long row = ((long)r * Tv + t) * Dv + n;
          const float gp   = gpre[row];
          const float gate = 1.0f / (1.0f + expf(-(gp + acc[nt][r])));
          const float o    = out[row] + s_attn[r][n] * gate;  // out holds mlp
          out[row]      = o;
          s_state[r][n] = o;
        }
      }
    }
    __syncthreads();
  }
}

// ----------------------------------------------------------------- launch ---
extern "C" void kernel_launch(void* const* d_in, const int* in_sizes, int n_in,
                              void* d_out, int out_size, void* d_ws, size_t ws_size,
                              hipStream_t stream) {
  const float* outputs = (const float*)d_in[0];
  const float* W_mlp   = (const float*)d_in[1];
  const float* b_mlp   = (const float*)d_in[2];
  const float* W_gate  = (const float*)d_in[3];
  const float* b_gate  = (const float*)d_in[4];
  float* out = (float*)d_out;

  char* ws = (char*)d_ws;                 // needs ~54 MB of scratch
  float*  gpre = (float*)(ws + WS_GPRE);
  __bf16* Wc   = (__bf16*)(ws + WS_WC);
  __bf16* Wa   = (__bf16*)(ws + WS_WA);
  __bf16* Abf  = (__bf16*)(ws + WS_ABF);

  pack_wc<<<(NTOT * KDIM) / 256, 256, 0, stream>>>(W_mlp, W_gate, Wc);
  pack_wa<<<(Dv * Dv) / 256, 256, 0, stream>>>(W_gate, Wa);
  pack_a <<<(Bv * Tv * Dv) / 256, 256, 0, stream>>>(outputs, Abf);

  dim3 grid1(NTOT / 64, (Bv * Tv) / 128);   // 16 N-tiles x 128 M-tiles
  mru_phase1<<<grid1, 256, 0, stream>>>(Abf, Wc, b_mlp, b_gate, out, gpre);
  mru_phase2<<<1, 512, 0, stream>>>(outputs, Wa, gpre, out);
}
